// MutiHeadAttention_86157043958275
// MI455X (gfx1250) — matrix operand order repaired and checked by reading, not compiled, not run on hardware
//
#include <hip/hip_runtime.h>
#include <cstdint>
#include <cstddef>

// Problem constants (match reference)
#define NH   16
#define DM   1024
#define DKV  64
#define BB   4
#define LL   1024
#define MTOK (BB * LL)   // 4096 rows of the model-dim matrices
#define KTILES 32        // K = 1024 -> 32 steps of 32

typedef __attribute__((ext_vector_type(16))) __bf16 v16bf;
typedef __attribute__((ext_vector_type(8)))  float  v8f;
typedef unsigned int v4u  __attribute__((ext_vector_type(4)));
typedef int          v8i_t __attribute__((ext_vector_type(8)));
typedef int          v4i_t __attribute__((ext_vector_type(4)));

// 16 bf16 halves = one WMMA A/B fragment per lane (two 16B chunks)
union Frag { uint4 u[2]; v16bf v; };

__device__ __forceinline__ uint16_t f2bf(float f) {
  uint32_t x = __float_as_uint(f);
  x += 0x7FFFu + ((x >> 16) & 1u);        // round-to-nearest-even
  return (uint16_t)(x >> 16);
}

__device__ __forceinline__ v8f vzero() {
  v8f z = {0.f, 0.f, 0.f, 0.f, 0.f, 0.f, 0.f, 0.f};
  return z;
}

// ---------------------------------------------------------------- fp32 -> bf16
__global__ __launch_bounds__(256) void cvt_f32_bf16(const float* __restrict__ in,
                                                    uint16_t* __restrict__ out, int n) {
  int i = blockIdx.x * blockDim.x + threadIdx.x;
  if (i < n) out[i] = f2bf(in[i]);
}

// -------------------------------------------- pack W[K][N] into B-fragment tiles
// kt-MAJOR tile order: tile (kt, nt) is 32(K) x 16(N) at linear tile index
// kt*NT + nt, so one K-step's B panel for 16 consecutive nt tiles is a single
// contiguous 16 KB chunk (TDM-friendly). Lane l owns column n = nt*16 + (l&15)
// and K rows (l>>4)*16 .. +15 of the 32-wide K window -> 16 contiguous halves.
__global__ __launch_bounds__(256) void pack_w(const float* __restrict__ W,
                                              uint16_t* __restrict__ P, int K, int N) {
  int p = blockIdx.x * blockDim.x + threadIdx.x;
  if (p >= K * N) return;
  int NT   = N >> 4;
  int j    = p & 15;
  int lane = (p >> 4) & 31;
  int t    = p >> 9;            // kt*NT + nt
  int nt   = t % NT;
  int kt   = t / NT;
  int kk   = kt * 32 + (lane >> 4) * 16 + j;
  int nn   = nt * 16 + (lane & 15);
  P[p] = f2bf(W[(size_t)kk * N + nn]);
}

// ------------------------------------------- WMMA GEMM with TDM-staged B panels
// Block = 128 threads (4 waves); block tile 32(M) x 256(N); wave tile 32 x 64.
// B panels (16 KB per K-step) are DMAed into a 2x16KB LDS double buffer by the
// Tensor Data Mover (wave 0 issues, distance-2 pipeline, s_wait_tensorcnt),
// fragments then come from LDS via ds_load_b128. A fragments load straight from
// row-major bf16 activations as b128 pairs.
// mode 0: store bf16 row-major [M x N]
// mode 1: store bf16 head-transposed: vt[((b*NH+head)*64+dv)*LL + token]
// mode 2: store f32 + residual add (for the Wo projection)
__global__ __launch_bounds__(128) void gemm_wmma_tdm(
    const uint16_t* __restrict__ A, const uint16_t* __restrict__ Bpk,
    int mode, uint16_t* __restrict__ obf, float* __restrict__ of32,
    const float* __restrict__ resid)
{
  __shared__ uint16_t bsm[2][8192];     // 2 x 16 KB
  const int NT = DM >> 4;               // 64 N-tiles total
  int lane = threadIdx.x & 31;
  int wave = threadIdx.x >> 5;
  int lrow = lane & 15, lh = lane >> 4;
  int m0  = blockIdx.y * 32;
  int ncb = blockIdx.x;                 // covers cols [ncb*256, ncb*256+256)
  int ntb = ncb * 16;

  const uint16_t* a0 = A + (size_t)(m0 + lrow) * DM;
  const uint16_t* a1 = A + (size_t)(m0 + 16 + lrow) * DM;

  // Tensor DMA descriptor: 1-D copy of 4096 dwords (16 KB) global -> LDS.
  // D# per CDNA5 ISA 8.3-8.5: group0 = {count|flags, lds_addr, gaddr_lo,
  // gaddr_hi|type=2}; group1 = {dsize=4B, tensor_dim0=4096, tensor_dim1=1,
  // tile_dim0=4096, tile_dim1=1, stride0=4096}; remaining groups zero (<=2D,
  // no gather / no iteration / no multicast).
  auto tdm_issue = [&](int kt, int buf) {
    uint32_t lds = (uint32_t)(size_t)&bsm[buf][0];   // generic->LDS offset (addr[31:0])
    uint64_t ga  = (uint64_t)(size_t)(Bpk + (size_t)(kt * NT + ntb) * 512);
    v4u g0;
    g0.x = 1u;                                        // count=1, user descriptor
    g0.y = lds;                                       // lds_addr
    g0.z = (uint32_t)ga;                              // global_addr[31:0]
    g0.w = (uint32_t)((ga >> 32) & 0x1FFFFFFu) | (2u << 30);  // [56:32] | type=2
    v8i_t g1;
    g1[0] = (int)(2u << 16);        // wg_mask=0, data_size=2 (4 bytes)
    g1[1] = (int)(4096u << 16);     // tensor_dim0[15:0] -> bits[63:48]
    g1[2] = (int)(1u << 16);        // tensor_dim0 hi=0 | tensor_dim1 lo=1
    g1[3] = (int)(4096u << 16);     // tensor_dim1 hi=0 | tile_dim0=4096
    g1[4] = 1;                      // tile_dim1=1, tile_dim2=0
    g1[5] = 4096;                   // tensor_dim0_stride lo
    g1[6] = 0;
    g1[7] = 0;
    v4i_t g2 = {0, 0, 0, 0};
    v4i_t g3 = {0, 0, 0, 0};
    v8i_t g4 = {0, 0, 0, 0, 0, 0, 0, 0};
    __builtin_amdgcn_tensor_load_to_lds(g0, g1, g2, g3, g4, 0);
  };

  if (wave == 0) { tdm_issue(0, 0); tdm_issue(1, 1); }

  v8f acc[2][4];
#pragma unroll
  for (int mi = 0; mi < 2; ++mi)
#pragma unroll
    for (int t = 0; t < 4; ++t) acc[mi][t] = vzero();

  for (int kt = 0; kt < KTILES; ++kt) {
    if (wave == 0) {
      if (kt == KTILES - 1) __builtin_amdgcn_s_wait_tensorcnt(0);
      else                  __builtin_amdgcn_s_wait_tensorcnt(1);
    }
    __syncthreads();                    // buf[kt&1] ready for every wave

    Frag a[2];
    a[0].u[0] = *(const uint4*)(a0 + lh * 16);
    a[0].u[1] = *(const uint4*)(a0 + lh * 16 + 8);
    a[1].u[0] = *(const uint4*)(a1 + lh * 16);
    a[1].u[1] = *(const uint4*)(a1 + lh * 16 + 8);
    __builtin_prefetch(a0 + 32, 0, 3);  // global_prefetch_b8 (next K tile)
    __builtin_prefetch(a1 + 32, 0, 3);

    const uint16_t* bbase = &bsm[kt & 1][0];
#pragma unroll
    for (int t = 0; t < 4; ++t) {
      const uint16_t* bp = bbase + ((wave * 4 + t) * 512 + lane * 16);
      Frag b;
      b.u[0] = *(const uint4*)(bp);          // ds_load_b128
      b.u[1] = *(const uint4*)(bp + 8);
      acc[0][t] = __builtin_amdgcn_wmma_f32_16x16x32_bf16(
          false, a[0].v, false, b.v, (short)0, acc[0][t], false, false);
      acc[1][t] = __builtin_amdgcn_wmma_f32_16x16x32_bf16(
          false, a[1].v, false, b.v, (short)0, acc[1][t], false, false);
    }
    a0 += 32; a1 += 32;

    __syncthreads();                    // all waves done reading buf[kt&1]
    if (wave == 0 && kt + 2 < KTILES) tdm_issue(kt + 2, kt & 1);
  }

  // Epilogue. C layout: VGPR r -> row r + 8*(lane>>4), col = lane&15 in tile.
#pragma unroll
  for (int mi = 0; mi < 2; ++mi) {
#pragma unroll
    for (int t = 0; t < 4; ++t) {
#pragma unroll
      for (int r = 0; r < 8; ++r) {
        int row = m0 + mi * 16 + r + 8 * lh;
        int col = ncb * 256 + wave * 64 + t * 16 + lrow;
        float val = acc[mi][t][r];
        if (mode == 0) {
          obf[(size_t)row * DM + col] = f2bf(val);
        } else if (mode == 1) {
          int head = col >> 6, dv = col & 63;
          int bb2 = row >> 10, tok = row & 1023;
          obf[((size_t)(bb2 * NH + head) * DKV + dv) * LL + tok] = f2bf(val);
        } else {
          of32[(size_t)row * DM + col] = val + resid[(size_t)row * DM + col];
        }
      }
    }
  }
}

// ----------------------------------------------------- flash attention (1 wave)
// Grid: (LL/16, NH, BB). qh/kh: [b, tok, n*64+d] bf16; vt: [b, n, dv, tok] bf16.
// Output aout in [b, n, q, dv] contiguous order (== reference's faithful .view).
__global__ __launch_bounds__(32) void attn_wmma(
    const uint16_t* __restrict__ qh, const uint16_t* __restrict__ kh,
    const uint16_t* __restrict__ vt, uint16_t* __restrict__ aout)
{
  __shared__ uint16_t pbuf[16 * 32];   // P tile bounce for C->A layout transpose
  int lane = threadIdx.x;
  int lrow = lane & 15, lh = lane >> 4;
  int q0 = blockIdx.x * 16;
  int n  = blockIdx.y;
  int b  = blockIdx.z;

  const uint16_t* Qp = qh + ((size_t)(b * LL + q0 + lrow) * DM) + n * DKV;
  Frag aq[2];
#pragma unroll
  for (int w = 0; w < 2; ++w) {
    aq[w].u[0] = *(const uint4*)(Qp + w * 32 + lh * 16);
    aq[w].u[1] = *(const uint4*)(Qp + w * 32 + lh * 16 + 8);
  }

  v8f o[4];
#pragma unroll
  for (int t = 0; t < 4; ++t) o[t] = vzero();
  float mrow[8], lsum[8];
#pragma unroll
  for (int r = 0; r < 8; ++r) { mrow[r] = -3.0e38f; lsum[r] = 0.f; }

  int ktiles = (q0 + 47) >> 5;         // causal: kv cols 0 .. q0+15
  for (int kv = 0; kv < ktiles; ++kv) {
    int k0 = kv * 32;
    v8f s[2]; s[0] = vzero(); s[1] = vzero();
#pragma unroll
    for (int hn = 0; hn < 2; ++hn) {
      const uint16_t* Kp =
          kh + ((size_t)(b * LL + k0 + hn * 16 + lrow) * DM) + n * DKV + lh * 16;
#pragma unroll
      for (int w = 0; w < 2; ++w) {
        Frag bk;
        bk.u[0] = *(const uint4*)(Kp + w * 32);
        bk.u[1] = *(const uint4*)(Kp + w * 32 + 8);
        s[hn] = __builtin_amdgcn_wmma_f32_16x16x32_bf16(
            false, aq[w].v, false, bk.v, (short)0, s[hn], false, false);
      }
    }

    // mask (before scale, faithful to reference), scale 1/8, online softmax
#pragma unroll
    for (int r = 0; r < 8; ++r) {
      int row = q0 + r + 8 * lh;
      float s0 = s[0][r]; if (k0 + lrow      > row) s0 = -1.0e9f;
      float s1 = s[1][r]; if (k0 + 16 + lrow > row) s1 = -1.0e9f;
      s0 *= 0.125f; s1 *= 0.125f;
      float tmax = fmaxf(s0, s1);
#pragma unroll
      for (int off = 1; off < 16; off <<= 1)
        tmax = fmaxf(tmax, __shfl_xor(tmax, off, 32));
      float nm = fmaxf(mrow[r], tmax);
      float sc = __expf(mrow[r] - nm);
      mrow[r] = nm;
      float p0 = __expf(s0 - nm);
      float p1 = __expf(s1 - nm);
      float ps = p0 + p1;
#pragma unroll
      for (int off = 1; off < 16; off <<= 1)
        ps += __shfl_xor(ps, off, 32);
      lsum[r] = lsum[r] * sc + ps;
#pragma unroll
      for (int t = 0; t < 4; ++t) o[t][r] *= sc;
      pbuf[(r + 8 * lh) * 32 + lrow]      = f2bf(p0);
      pbuf[(r + 8 * lh) * 32 + 16 + lrow] = f2bf(p1);
    }
    asm volatile("s_wait_dscnt 0x0" ::: "memory");
    Frag ap;
    ap.u[0] = *(const uint4*)&pbuf[lrow * 32 + lh * 16];
    ap.u[1] = *(const uint4*)&pbuf[lrow * 32 + lh * 16 + 8];

    // O += P @ V (V head-transposed: contiguous in kv per dv row)
#pragma unroll
    for (int t = 0; t < 4; ++t) {
      const uint16_t* Vp =
          vt + ((size_t)((b * NH + n) * DKV + t * 16 + lrow) * LL) + k0 + lh * 16;
      Frag bv;
      bv.u[0] = *(const uint4*)(Vp);
      bv.u[1] = *(const uint4*)(Vp + 8);
      o[t] = __builtin_amdgcn_wmma_f32_16x16x32_bf16(
          false, ap.v, false, bv.v, (short)0, o[t], false, false);
    }
  }

#pragma unroll
  for (int r = 0; r < 8; ++r) {
    float inv = 1.0f / lsum[r];
    int row = q0 + r + 8 * lh;
#pragma unroll
    for (int t = 0; t < 4; ++t) {
      aout[((size_t)(b * NH + n) * LL + row) * DKV + t * 16 + lrow] =
          f2bf(o[t][r] * inv);
    }
  }
}

// ------------------------------------------------------------------- LayerNorm
__global__ __launch_bounds__(256) void ln_kernel(
    const float* __restrict__ y, const float* __restrict__ gamma,
    const float* __restrict__ beta, float* __restrict__ out)
{
  __shared__ float s1[8], s2[8];
  int row = blockIdx.x;
  int tid = threadIdx.x;
  const float4* yr = (const float4*)(y + (size_t)row * DM);
  float4 x = yr[tid];
  float sum = x.x + x.y + x.z + x.w;
  float ssq = x.x * x.x + x.y * x.y + x.z * x.z + x.w * x.w;
#pragma unroll
  for (int off = 1; off < 32; off <<= 1) {
    sum += __shfl_xor(sum, off, 32);
    ssq += __shfl_xor(ssq, off, 32);
  }
  int lane = tid & 31, wv = tid >> 5;
  if (lane == 0) { s1[wv] = sum; s2[wv] = ssq; }
  __syncthreads();
  float ts = 0.f, tq = 0.f;
#pragma unroll
  for (int i = 0; i < 8; ++i) { ts += s1[i]; tq += s2[i]; }
  float mu   = ts * (1.0f / DM);
  float var  = tq * (1.0f / DM) - mu * mu;
  float rstd = rsqrtf(var + 1e-6f);
  const float4* g4 = (const float4*)gamma;
  const float4* b4 = (const float4*)beta;
  float4 g = g4[tid], bt = b4[tid];
  float4 o;
  o.x = (x.x - mu) * rstd * g.x + bt.x;
  o.y = (x.y - mu) * rstd * g.y + bt.y;
  o.z = (x.z - mu) * rstd * g.z + bt.z;
  o.w = (x.w - mu) * rstd * g.w + bt.w;
  ((float4*)(out + (size_t)row * DM))[tid] = o;
}

// ------------------------------------------------------------------- launcher
extern "C" void kernel_launch(void* const* d_in, const int* in_sizes, int n_in,
                              void* d_out, int out_size, void* d_ws, size_t ws_size,
                              hipStream_t stream) {
  (void)in_sizes; (void)n_in; (void)out_size; (void)ws_size;
  const float* q     = (const float*)d_in[0];
  const float* k     = (const float*)d_in[1];
  const float* v     = (const float*)d_in[2];
  // d_in[3] = mask (causal computed analytically)
  const float* Wq    = (const float*)d_in[4];
  const float* Wk    = (const float*)d_in[5];
  const float* Wv    = (const float*)d_in[6];
  const float* Wo    = (const float*)d_in[7];
  const float* gamma = (const float*)d_in[8];
  const float* beta  = (const float*)d_in[9];
  float* out = (float*)d_out;

  char* wsb = (char*)d_ws;
  size_t off = 0;
  auto take = [&](size_t bytes) -> void* {
    void* p = wsb + off;
    off += (bytes + 255) & ~(size_t)255;
    return p;
  };
  const size_t actN = (size_t)MTOK * DM;                 // 4M elements
  uint16_t* xq  = (uint16_t*)take(actN * 2);
  uint16_t* xk  = (uint16_t*)take(actN * 2);
  uint16_t* xv  = (uint16_t*)take(actN * 2);
  uint16_t* pWq = (uint16_t*)take((size_t)DM * DM * 2);
  uint16_t* pWk = (uint16_t*)take((size_t)DM * DM * 2);
  uint16_t* pWv = (uint16_t*)take((size_t)DM * DM * 2);
  uint16_t* pWo = (uint16_t*)take((size_t)DM * DM * 2);
  uint16_t* qhb = (uint16_t*)take(actN * 2);
  uint16_t* khb = (uint16_t*)take(actN * 2);
  uint16_t* vtb = (uint16_t*)take(actN * 2);
  uint16_t* aob = (uint16_t*)take(actN * 2);
  float*    yb  = (float*)take(actN * 4);

  const int nact = (int)actN;
  cvt_f32_bf16<<<(nact + 255) / 256, 256, 0, stream>>>(q, xq, nact);
  cvt_f32_bf16<<<(nact + 255) / 256, 256, 0, stream>>>(k, xk, nact);
  cvt_f32_bf16<<<(nact + 255) / 256, 256, 0, stream>>>(v, xv, nact);

  const int nw = DM * DM;
  pack_w<<<(nw + 255) / 256, 256, 0, stream>>>(Wq, pWq, DM, DM);
  pack_w<<<(nw + 255) / 256, 256, 0, stream>>>(Wk, pWk, DM, DM);
  pack_w<<<(nw + 255) / 256, 256, 0, stream>>>(Wv, pWv, DM, DM);
  pack_w<<<(nw + 255) / 256, 256, 0, stream>>>(Wo, pWo, DM, DM);

  dim3 gg(DM / 256, MTOK / 32);  // (4, 128), 128 threads = 4 waves
  gemm_wmma_tdm<<<gg, 128, 0, stream>>>(xq, pWq, 0, qhb, nullptr, nullptr);
  gemm_wmma_tdm<<<gg, 128, 0, stream>>>(xk, pWk, 0, khb, nullptr, nullptr);
  gemm_wmma_tdm<<<gg, 128, 0, stream>>>(xv, pWv, 1, vtb, nullptr, nullptr);

  attn_wmma<<<dim3(LL / 16, NH, BB), 32, 0, stream>>>(qhb, khb, vtb, aob);

  gemm_wmma_tdm<<<gg, 128, 0, stream>>>(aob, pWo, 2, nullptr, yb, q);
  ln_kernel<<<MTOK, 256, 0, stream>>>(yb, gamma, beta, out);
}
